// TransMIL_7215545057828
// MI455X (gfx1250) — matrix operand order repaired and checked by reading, not compiled, not run on hardware
//
#include <hip/hip_runtime.h>
#include <math.h>

// ---------------------------------------------------------------------------
// MI455X / gfx1250, wave32. All GEMM work on v_wmma_f32_16x16x32_bf16.
// K-tiles in the flash kernel are moved by the Tensor Data Mover when the
// toolchain exposes __builtin_amdgcn_tensor_load_to_lds.
// ---------------------------------------------------------------------------

typedef __attribute__((ext_vector_type(16))) __bf16 bf16x16;
typedef __attribute__((ext_vector_type(8)))  __bf16 bf16x8;
typedef __attribute__((ext_vector_type(4)))  __bf16 bf16x4;
typedef __attribute__((ext_vector_type(8)))  float  f32x8;
typedef __attribute__((ext_vector_type(4)))  float  f32x4;
typedef __attribute__((ext_vector_type(4)))  unsigned int u32x4;
typedef __attribute__((ext_vector_type(8)))  int i32x8;
typedef __attribute__((ext_vector_type(4)))  int i32x4;

#define DEV __device__ __forceinline__

#if defined(__HIP_DEVICE_COMPILE__) && __has_builtin(__builtin_amdgcn_tensor_load_to_lds)
#define HAVE_TDM 1
#else
#define HAVE_TDM 0
#endif

DEV f32x8 wmma_bf16(bf16x16 a, bf16x16 b, f32x8 c){
#if defined(__HIP_DEVICE_COMPILE__)
  return __builtin_amdgcn_wmma_f32_16x16x32_bf16(false, a, false, b, (short)0, c, false, false);
#else
  return c;
#endif
}

DEV f32x8 zero8(){ f32x8 z = {0.f,0.f,0.f,0.f,0.f,0.f,0.f,0.f}; return z; }

DEV bf16x4 cvt4(f32x4 v){
  bf16x4 r; r[0]=(__bf16)v[0]; r[1]=(__bf16)v[1]; r[2]=(__bf16)v[2]; r[3]=(__bf16)v[3]; return r;
}
DEV f32x4 b2f4(bf16x4 h){
  f32x4 r; r[0]=(float)h[0]; r[1]=(float)h[1]; r[2]=(float)h[2]; r[3]=(float)h[3]; return r;
}

// ---------------------------------------------------------------------------
// Tensor Data Mover: 2D tile of bf16 rows -> LDS (packed row-major).
// D# layout per CDNA5 ISA ch.8 (group0: count/lds/global/type, group1 dims).
// tile: rows x rowElems elements, tensor row stride rowStrideElems.
// ---------------------------------------------------------------------------
DEV void tdm_load_2d_bf16(unsigned ldsOff, const void* gptr, int rows,
                          int rowElems, int rowStrideElems){
#if HAVE_TDM
  unsigned long long ga = (unsigned long long)(uintptr_t)gptr;
  u32x4 g0;
  g0[0] = 1u;                                                  // count=1
  g0[1] = ldsOff;                                              // lds_addr
  g0[2] = (unsigned)(ga & 0xffffffffu);                        // global_addr lo
  g0[3] = (unsigned)((ga >> 32) & 0x01ffffffu) | 0x80000000u;  // ga hi | type=2
  i32x8 g1;
  g1[0] = (1<<16);                              // data_size = 1 (2 bytes)
  g1[1] = (rowElems & 0xffff) << 16;            // tensor_dim0[15:0]  @ bits 63:48
  g1[2] = (rows & 0xffff) << 16;                // tensor_dim1[15:0]  @ bits 95:80
  g1[3] = (rowElems & 0xffff) << 16;            // tile_dim0          @ bits 127:112
  g1[4] = (rows & 0xffff);                      // tile_dim1          @ bits 143:128
  g1[5] = rowStrideElems;                       // tensor_dim0_stride lo32
  g1[6] = 0; g1[7] = 0;
  i32x4 gz = {0,0,0,0};
#if __clang_major__ >= 23
  i32x8 gz8 = {0,0,0,0,0,0,0,0};
  __builtin_amdgcn_tensor_load_to_lds(g0, g1, gz, gz, gz8, 0);
#else
  __builtin_amdgcn_tensor_load_to_lds(g0, g1, gz, gz, 0);
#endif
#else
  (void)ldsOff; (void)gptr; (void)rows; (void)rowElems; (void)rowStrideElems;
#endif
}

DEV void tdm_wait(){
#if HAVE_TDM
  __builtin_amdgcn_s_wait_tensorcnt(0);
#endif
}

// A fragment (16x32 tile) from row-major LDS, stride ld (elements).
// lanes 0-15: row=lane, elems 0..7 -> K 0..7, elems 8..15 -> K 16..23
// lanes 16-31: row=lane-16, elems 0..7 -> K 8..15, elems 8..15 -> K 24..31
DEV bf16x16 ldsAfrag(const __bf16* base, int ld){
  int lane = threadIdx.x & 31; int hi = lane >> 4, ln = lane & 15;
  const __bf16* p = base + ln*ld;
  bf16x16 a;
#pragma unroll
  for (int e=0;e<8;e++){ a[e] = p[hi*8+e]; a[8+e] = p[16+hi*8+e]; }
  return a;
}

// B fragment (32x16 tile) from LDS stored as Bt[n][k] (n-major), stride ld.
// lanes 0-15: col=lane, elems 0..7 -> K 0..7, elems 8..15 -> K 16..23
// lanes 16-31: col=lane-16, elems 0..7 -> K 8..15, elems 8..15 -> K 24..31
DEV bf16x16 ldsBfragT(const __bf16* base, int ld){
  int lane = threadIdx.x & 31; int hi = lane >> 4, ln = lane & 15;
  const __bf16* p = base + ln*ld;
  bf16x16 b;
#pragma unroll
  for (int e=0;e<8;e++){ b[e] = p[hi*8+e]; b[8+e] = p[16+hi*8+e]; }
  return b;
}

// ------------------------- block reductions (256 thr) ----------------------
DEV float blockReduceSum(float v){
  __shared__ float sh[8];
  int lane = threadIdx.x & 31, wv = threadIdx.x >> 5;
#pragma unroll
  for (int o=16;o;o>>=1) v += __shfl_xor(v,o);
  __syncthreads();
  if (lane==0) sh[wv]=v;
  __syncthreads();
  float r = (threadIdx.x<8)? sh[threadIdx.x] : 0.f;
  if (wv==0){
#pragma unroll
    for (int o=4;o;o>>=1) r += __shfl_xor(r,o);
    if (lane==0) sh[0]=r;
  }
  __syncthreads();
  return sh[0];
}

DEV float blockReduceMax(float v){
  __shared__ float sh[8];
  int lane = threadIdx.x & 31, wv = threadIdx.x >> 5;
#pragma unroll
  for (int o=16;o;o>>=1) v = fmaxf(v, __shfl_xor(v,o));
  __syncthreads();
  if (lane==0) sh[wv]=v;
  __syncthreads();
  float r = (threadIdx.x<8)? sh[threadIdx.x] : -1e30f;
  if (wv==0){
#pragma unroll
    for (int o=4;o;o>>=1) r = fmaxf(r, __shfl_xor(r,o));
    if (lane==0) sh[0]=r;
  }
  __syncthreads();
  return sh[0];
}

// ===========================================================================
// Big GEMM: C = A(MxK) @ W(KxN). Block tile 128x128, K-step 32, 8 waves of
// 32x64, register double-buffered global->LDS staging, bf16 WMMA.
// MODE 0: A f32 (x), epilogue bias+GELU(tanh) -> hbuf (+1 cls row offset)
// MODE 1: A bf16 (xpad), epilogue split to q(*0.125)/k/v bf16 head-major
// MODE 2: A bf16 (xo, row-mapped past 255 pad rows, clamped), hres += C+bias
// ===========================================================================
template<int MODE>
__global__ __launch_bounds__(256) void gemm_big(
    const float* __restrict__ Af, const __bf16* __restrict__ Ab,
    const float* __restrict__ W, const float* __restrict__ bias,
    int M, int K, int Nc, float* __restrict__ outF,
    __bf16* __restrict__ qOut, __bf16* __restrict__ kOut, __bf16* __restrict__ vOut)
{
  __shared__ __bf16 As[128*40];   // [row][k]
  __shared__ __bf16 Bs[128*40];   // [n][k] transposed for contiguous frags
  int m0 = blockIdx.x*128, n0 = blockIdx.y*128;
  int tid = threadIdx.x, wave = tid>>5, lane = tid&31, hi = lane>>4, ln = lane&15;
  int rg = wave>>1, cg = wave&1;

  f32x8 acc[2][4];
#pragma unroll
  for (int rt=0;rt<2;rt++)
#pragma unroll
    for (int ct=0;ct<4;ct++) acc[rt][ct] = zero8();

  f32x4 aRf[4]; u32x4 aRb[2]; f32x4 wR[4];

  auto loadA = [&](int kk){
    if (MODE==0){
#pragma unroll
      for (int i=0;i<4;i++){
        int c = i*256+tid; int row = c>>3, off = (c&7)*4;
        aRf[i] = *(const f32x4*)&Af[(size_t)(m0+row)*K + kk + off];
      }
    } else {
#pragma unroll
      for (int i=0;i<2;i++){
        int c = i*256+tid; int row = c>>2, off = (c&3)*8;
        int gr = m0 + row;
        size_t sr;
        if (MODE==1) sr = (size_t)gr;
        else {
          int gcl = gr < M ? gr : (M-1);                  // branchless clamp
          int bb = gcl/16385, ii = gcl%16385;
          sr = (size_t)(bb*16640 + 255 + ii);
        }
        aRb[i] = *(const u32x4*)&Ab[sr*K + kk + off];
      }
    }
  };
  auto loadW = [&](int kk){
#pragma unroll
    for (int i=0;i<4;i++){
      int c = i*256+tid; int row = c>>5, col = (c&31)*4;
      wR[i] = *(const f32x4*)&W[(size_t)(kk+row)*Nc + n0 + col];
    }
  };
  auto stage = [&](){
    if (MODE==0){
#pragma unroll
      for (int i=0;i<4;i++){
        int c = i*256+tid; int row = c>>3, off = (c&7)*4;
        *(bf16x4*)&As[row*40+off] = cvt4(aRf[i]);
      }
    } else {
#pragma unroll
      for (int i=0;i<2;i++){
        int c = i*256+tid; int row = c>>2, off = (c&3)*8;
        *(u32x4*)&As[row*40+off] = aRb[i];
      }
    }
#pragma unroll
    for (int i=0;i<4;i++){
      int c = i*256+tid; int row = c>>5, col = (c&31)*4;
      bf16x4 b4 = cvt4(wR[i]);
#pragma unroll
      for (int j=0;j<4;j++) Bs[(col+j)*40+row] = b4[j];
    }
  };

  loadA(0); loadW(0);
  for (int kk=0; kk<K; kk+=32){
    __syncthreads();
    stage();
    if (kk+32 < K){ loadA(kk+32); loadW(kk+32); }   // overlap with compute
    __syncthreads();

    bf16x16 a0 = ldsAfrag(As + (rg*32   )*40, 40);
    bf16x16 a1 = ldsAfrag(As + (rg*32+16)*40, 40);
#pragma unroll
    for (int ct=0;ct<4;ct++){
      bf16x16 b = ldsBfragT(Bs + (cg*64 + ct*16)*40, 40);
      acc[0][ct] = wmma_bf16(a0, b, acc[0][ct]);
      acc[1][ct] = wmma_bf16(a1, b, acc[1][ct]);
    }
  }

#pragma unroll
  for (int rt=0;rt<2;rt++)
#pragma unroll
    for (int ct=0;ct<4;ct++)
#pragma unroll
      for (int vv=0; vv<8; vv++){
        int gr = m0 + rg*32 + rt*16 + hi*8 + vv;
        int gc = n0 + cg*64 + ct*16 + ln;
        if (gr >= M) continue;
        float c = acc[rt][ct][vv];
        if (MODE==0){
          c += bias[gc];
          c = 0.5f*c*(1.f + tanhf(0.7978845608f*(c + 0.044715f*c*c*c)));
          int bb = gr>>14, tt = gr & 16383;
          outF[((size_t)(bb*16385) + 1 + tt)*512 + gc] = c;
        } else if (MODE==1){
          int bb = gr/16640, tt = gr%16640;
          int which = gc>>9, dd = gc&511, hh = dd>>6, dh = dd&63;
          size_t di = (((size_t)(bb*8+hh))*16640 + tt)*64 + dh;
          if (which==0)      qOut[di] = (__bf16)(c*0.125f);
          else if (which==1) kOut[di] = (__bf16)c;
          else               vOut[di] = (__bf16)c;
        } else {
          int bb = gr/16385, ii = gr%16385;
          outF[((size_t)(bb*16385) + ii)*512 + gc] += c + bias[gc];
        }
      }
}

// ===========================================================================
// Batched small GEMM with bf16x3 compensation (~fp32 accuracy on WMMA):
//   OUT = s0*(A@B[^T]) + s1*I + s2*Cadd,  optional bf16 copy of OUT.
// A: f32 (M x K), B: f32 (K x N) or (N x K) if TRANSB. 64x64 tiles.
// ===========================================================================
template<bool TRANSB>
__global__ __launch_bounds__(256) void gemm_small_x3(
    const float* __restrict__ A, const float* __restrict__ B,
    float* __restrict__ C, __bf16* __restrict__ Cbf, const float* __restrict__ Cadd,
    int M, int N, int K, long sA, long sB, long sC,
    float s0, float s1, float s2)
{
  int bat = blockIdx.z;
  const float* Abat = A + (size_t)bat*sA;
  const float* Bbat = B + (size_t)bat*sB;
  float*       Cbat = C + (size_t)bat*sC;
  const float* Caddb = Cadd ? (Cadd + (size_t)bat*sC) : nullptr;
  int m0 = blockIdx.x*64, n0 = blockIdx.y*64;

  __shared__ __bf16 Ah[64*40], Al[64*40];   // [row][k] hi/lo
  __shared__ __bf16 Bh[64*40], Bl[64*40];   // [n][k]   hi/lo
  int tid = threadIdx.x, wave = tid>>5, lane = tid&31, hi = lane>>4, ln = lane&15;
  int rt = wave>>1, cp = wave&1;            // 4 row groups x 2 col pairs

  f32x8 acc[2]; acc[0] = zero8(); acc[1] = zero8();

  f32x4 aR[2], bR[2];
  auto loadA = [&](int kk){
#pragma unroll
    for (int i=0;i<2;i++){
      int c = i*256+tid; int row = c>>3, off = (c&7)*4;
      aR[i] = *(const f32x4*)&Abat[(size_t)(m0+row)*K + kk + off];
    }
  };
  auto loadB = [&](int kk){
#pragma unroll
    for (int i=0;i<2;i++){
      int c = i*256+tid;
      if (!TRANSB){
        int row = c>>4, col = (c&15)*4;                 // B 32 x 64
        bR[i] = *(const f32x4*)&Bbat[(size_t)(kk+row)*N + n0 + col];
      } else {
        int row = c>>3, off = (c&7)*4;                  // B^T rows: 64 x 32
        bR[i] = *(const f32x4*)&Bbat[(size_t)(n0+row)*K + kk + off];
      }
    }
  };
  auto stage = [&](){
#pragma unroll
    for (int i=0;i<2;i++){
      int c = i*256+tid; int row = c>>3, off = (c&7)*4;
      bf16x4 h = cvt4(aR[i]);
      bf16x4 l = cvt4(aR[i] - b2f4(h));
      *(bf16x4*)&Ah[row*40+off] = h;
      *(bf16x4*)&Al[row*40+off] = l;
    }
#pragma unroll
    for (int i=0;i<2;i++){
      int c = i*256+tid;
      bf16x4 h = cvt4(bR[i]);
      bf16x4 l = cvt4(bR[i] - b2f4(h));
      if (!TRANSB){
        int row = c>>4, col = (c&15)*4;
#pragma unroll
        for (int j=0;j<4;j++){ Bh[(col+j)*40+row] = h[j]; Bl[(col+j)*40+row] = l[j]; }
      } else {
        int row = c>>3, off = (c&7)*4;
        *(bf16x4*)&Bh[row*40+off] = h;
        *(bf16x4*)&Bl[row*40+off] = l;
      }
    }
  };

  loadA(0); loadB(0);
  for (int kk=0; kk<K; kk+=32){
    __syncthreads();
    stage();
    if (kk+32 < K){ loadA(kk+32); loadB(kk+32); }
    __syncthreads();

    bf16x16 ah = ldsAfrag(Ah + rt*16*40, 40);
    bf16x16 al = ldsAfrag(Al + rt*16*40, 40);
#pragma unroll
    for (int cc=0; cc<2; cc++){
      int ct = cp*2 + cc;
      bf16x16 bh = ldsBfragT(Bh + (ct*16)*40, 40);
      bf16x16 bl = ldsBfragT(Bl + (ct*16)*40, 40);
      f32x8 c = acc[cc];
      c = wmma_bf16(ah, bh, c);
      c = wmma_bf16(ah, bl, c);
      c = wmma_bf16(al, bh, c);
      acc[cc] = c;
    }
  }

#pragma unroll
  for (int cc=0; cc<2; cc++){
#pragma unroll
    for (int vv=0; vv<8; vv++){
      int r  = m0 + rt*16 + hi*8 + vv;
      int cl = n0 + (cp*2+cc)*16 + ln;
      float val = s0*acc[cc][vv];
      if (s1 != 0.f && r == cl) val += s1;
      if (Caddb) val += s2*Caddb[(size_t)r*N + cl];
      Cbat[(size_t)r*N + cl] = val;
      if (Cbf) Cbf[(size_t)bat*sC + (size_t)r*N + cl] = (__bf16)val;
    }
  }
}

// ===========================================================================
// Flash kernel: OUT = softmax(Q @ K^T) @ V, online softmax, key-chunked.
// Q:(bh,nRows,64) K,V:(bh,nKeys,64) bf16.  Block = 256 thr, 8 waves x 32 rows.
// K-tile staged by the Tensor Data Mover (wave 0 issues, s_wait_tensorcnt).
// nChunks==1 -> write normalized out f32; else write (m,l,acc) partials.
// ===========================================================================
__global__ __launch_bounds__(256) void flash_kernel(
    const __bf16* __restrict__ Q, const __bf16* __restrict__ Km, const __bf16* __restrict__ Vm,
    int nRows, int nKeys, int nChunks,
    float* __restrict__ outAcc, float* __restrict__ mPart, float* __restrict__ lPart)
{
  int bh = blockIdx.z, chunk = blockIdx.y;
  int kpc = nKeys / nChunks;
  int keyBase = chunk * kpc;
  const __bf16* Qb = Q  + (size_t)bh*nRows*64;
  const __bf16* Kb = Km + (size_t)bh*nKeys*64;
  const __bf16* Vb = Vm + (size_t)bh*nKeys*64;

  int tid = threadIdx.x, wave = tid>>5, lane = tid&31, hi = lane>>4, ln = lane&15;
  int wr = blockIdx.x*256 + wave*32;

  __shared__ __bf16 Kt[32*64];        // [key][d] (packed row-major, TDM dest)
  __shared__ __bf16 Vtt[64*40];       // [d][k] transposed
  __shared__ __bf16 Pt[8][32*32];     // per-wave P staging

  unsigned ktOff = (unsigned)(uintptr_t)(&Kt[0]);   // LDS byte offset

  bf16x16 afr[2][2];
#pragma unroll
  for (int rt=0; rt<2; rt++){
    const __bf16* qp = Qb + (size_t)(wr + rt*16 + ln)*64;
#pragma unroll
    for (int ks=0; ks<2; ks++){
      bf16x16 a; int kb = ks*32 + hi*8;
#pragma unroll
      for (int e=0;e<8;e++){ a[e] = qp[kb+e]; a[8+e] = qp[kb+16+e]; }
      afr[rt][ks] = a;
    }
  }

  float m[16], l[16];
#pragma unroll
  for (int j=0;j<16;j++){ m[j] = -1e30f; l[j] = 0.f; }
  f32x8 acc[2][4];
#pragma unroll
  for (int rt=0;rt<2;rt++){ acc[rt][0]=zero8(); acc[rt][1]=zero8(); acc[rt][2]=zero8(); acc[rt][3]=zero8(); }

  int ldRow = tid>>3, ldOff = (tid&7)*8;            // 256 chunks of 8 bf16
  for (int kk=0; kk<kpc; kk+=32){
    __syncthreads();
#if HAVE_TDM
    if (tid < 32)                                   // wave 0 drives the TDM
      tdm_load_2d_bf16(ktOff, &Kb[(size_t)(keyBase+kk)*64], 32, 64, 64);
#endif
    {
      size_t srow = (size_t)(keyBase + kk + ldRow)*64 + ldOff;
      bf16x8 vv8 = *(const bf16x8*)&Vb[srow];
#if !HAVE_TDM
      bf16x8 kv = *(const bf16x8*)&Kb[srow];
      *(bf16x8*)&Kt[ldRow*64 + ldOff] = kv;
#endif
#pragma unroll
      for (int e=0;e<8;e++) Vtt[(ldOff+e)*40 + ldRow] = vv8[e];
      if (kk + 32 < kpc)
        __builtin_prefetch(&Vb[srow + 32*64], 0, 1);
    }
#if HAVE_TDM
    if (tid < 32) tdm_wait();
#endif
    __syncthreads();

    f32x8 S[2][2];
#pragma unroll
    for (int rt=0;rt<2;rt++)
#pragma unroll
      for (int kt=0;kt<2;kt++){
        f32x8 c = zero8();
#pragma unroll
        for (int ks=0;ks<2;ks++){
          bf16x16 b = ldsBfragT(Kt + (kt*16)*64 + ks*32, 64);
          c = wmma_bf16(afr[rt][ks], b, c);
        }
        S[rt][kt] = c;
      }

#pragma unroll
    for (int rt=0;rt<2;rt++){
#pragma unroll
      for (int vv=0; vv<8; vv++){
        float x = fmaxf(S[rt][0][vv], S[rt][1][vv]);
        x = fmaxf(x, __shfl_xor(x,1)); x = fmaxf(x, __shfl_xor(x,2));
        x = fmaxf(x, __shfl_xor(x,4)); x = fmaxf(x, __shfl_xor(x,8));
        int j = rt*8 + vv;
        float mn = fmaxf(m[j], x);
        float sc = __expf(m[j] - mn);
        float p0 = __expf(S[rt][0][vv] - mn);
        float p1 = __expf(S[rt][1][vv] - mn);
        float ps = p0 + p1;
        ps += __shfl_xor(ps,1); ps += __shfl_xor(ps,2);
        ps += __shfl_xor(ps,4); ps += __shfl_xor(ps,8);
        l[j] = l[j]*sc + ps;
        m[j] = mn;
#pragma unroll
        for (int ct=0;ct<4;ct++) acc[rt][ct][vv] *= sc;
        int prow = rt*16 + hi*8 + vv;
        Pt[wave][prow*32 + ln]      = (__bf16)p0;
        Pt[wave][prow*32 + 16 + ln] = (__bf16)p1;
      }
    }
    __syncthreads();

#pragma unroll
    for (int rt=0;rt<2;rt++){
      bf16x16 a = ldsAfrag(&Pt[wave][rt*16*32], 32);
#pragma unroll
      for (int ct=0;ct<4;ct++){
        bf16x16 b = ldsBfragT(Vtt + (ct*16)*40, 40);
        acc[rt][ct] = wmma_bf16(a, b, acc[rt][ct]);
      }
    }
  }

  if (nChunks == 1){
    float* ob = outAcc + (size_t)bh*nRows*64;
#pragma unroll
    for (int rt=0;rt<2;rt++)
#pragma unroll
      for (int vv=0; vv<8; vv++){
        int row = wr + rt*16 + hi*8 + vv;
        float inv = 1.f / l[rt*8+vv];
#pragma unroll
        for (int ct=0;ct<4;ct++)
          ob[(size_t)row*64 + ct*16 + ln] = acc[rt][ct][vv]*inv;
      }
  } else {
    size_t base = ((size_t)bh*nChunks + chunk)*nRows;
    float* ab = outAcc + base*64;
#pragma unroll
    for (int rt=0;rt<2;rt++)
#pragma unroll
      for (int vv=0; vv<8; vv++){
        int row = wr + rt*16 + hi*8 + vv;
        int j = rt*8 + vv;
        if (ln == 0){ mPart[base+row] = m[j]; lPart[base+row] = l[j]; }
#pragma unroll
        for (int ct=0;ct<4;ct++)
          ab[(size_t)row*64 + ct*16 + ln] = acc[rt][ct][vv];
      }
  }
}

__global__ void flash_merge(const float* __restrict__ accp, const float* __restrict__ mp,
                            const float* __restrict__ lp, float* __restrict__ av,
                            int nRows, int nChunks)
{
  size_t idx = (size_t)blockIdx.x*blockDim.x + threadIdx.x;
  size_t total = (size_t)16*nRows*64;
  if (idx >= total) return;
  int d = idx & 63;
  size_t rw = idx >> 6;
  int row = (int)(rw % nRows); int bh = (int)(rw / nRows);
  float gm = -1e30f;
  for (int c=0;c<nChunks;c++)
    gm = fmaxf(gm, mp[((size_t)bh*nChunks+c)*nRows + row]);
  float L = 0.f, A = 0.f;
  for (int c=0;c<nChunks;c++){
    size_t b = ((size_t)bh*nChunks+c)*nRows + row;
    float w = __expf(mp[b]-gm);
    L += lp[b]*w;
    A += accp[b*64 + d]*w;
  }
  av[idx] = A / L;
}

// ------------------------------- small kernels ------------------------------
__global__ void cls_init(const float* __restrict__ cls, float* __restrict__ h){
  int idx = blockIdx.x*blockDim.x + threadIdx.x;
  if (idx >= 1024) return;
  int b = idx>>9, c = idx&511;
  h[(size_t)b*16385*512 + c] = cls[c];
}

__global__ __launch_bounds__(256) void ln_to_xpad(
    const float* __restrict__ h, const float* __restrict__ g,
    const float* __restrict__ beta, __bf16* __restrict__ xpad)
{
  int row = blockIdx.x;                       // 0..33279
  int bt = row / 16640, t = row % 16640;
  __bf16* dst = xpad + (size_t)row*512;
  if (t < 255){
    for (int c=threadIdx.x; c<512; c+=256) dst[c] = (__bf16)0.f;
    return;
  }
  const float* src = h + ((size_t)(bt*16385) + (t-255))*512;
  float s = 0.f;
  for (int c=threadIdx.x; c<512; c+=256) s += src[c];
  float mu = blockReduceSum(s) * (1.f/512.f);
  float v = 0.f;
  for (int c=threadIdx.x; c<512; c+=256){ float d = src[c]-mu; v += d*d; }
  float var = blockReduceSum(v) * (1.f/512.f);
  float rstd = rsqrtf(var + 1e-5f);
  for (int c=threadIdx.x; c<512; c+=256)
    dst[c] = (__bf16)((src[c]-mu)*rstd*g[c] + beta[c]);
}

__global__ void landmarks(const __bf16* __restrict__ q, const __bf16* __restrict__ k,
                          float* __restrict__ qlf, float* __restrict__ klf,
                          __bf16* __restrict__ qlb, __bf16* __restrict__ klb)
{
  int idx = blockIdx.x*blockDim.x + threadIdx.x;   // 16*256*64
  if (idx >= 16*256*64) return;
  int d = idx & 63, mm = (idx>>6) & 255, bh = idx>>14;
  size_t base = ((size_t)bh*16640 + mm*65)*64 + d;
  float sq = 0.f, sk = 0.f;
  for (int j=0;j<65;j++){ sq += (float)q[base + (size_t)j*64]; sk += (float)k[base + (size_t)j*64]; }
  sq *= (1.f/65.f); sk *= (1.f/65.f);
  qlf[idx] = sq; klf[idx] = sk; qlb[idx] = (__bf16)sq; klb[idx] = (__bf16)sk;
}

__global__ __launch_bounds__(256) void softmax256(float* __restrict__ x){
  size_t row = blockIdx.x;
  float* p = x + row*256;
  float v = p[threadIdx.x];
  float mx = blockReduceMax(v);
  float e = __expf(v - mx);
  float s = blockReduceSum(e);
  p[threadIdx.x] = e / s;
}

__global__ void colrow_sums(const float* __restrict__ a2, float* __restrict__ rs, float* __restrict__ cs){
  int i = threadIdx.x, bh = blockIdx.x;
  const float* p = a2 + ((size_t)bh<<16);
  float r = 0.f, c = 0.f;
  for (int j=0;j<256;j++){ r += fabsf(p[i*256+j]); c += fabsf(p[j*256+i]); }
  rs[bh*256+i] = r; cs[bh*256+i] = c;
}

__global__ __launch_bounds__(256) void max_scale(const float* __restrict__ rs, const float* __restrict__ cs,
                                                 float* __restrict__ scale){
  float mr = -1e30f, mc = -1e30f;
  for (int i=threadIdx.x; i<4096; i+=256){ mr = fmaxf(mr, rs[i]); mc = fmaxf(mc, cs[i]); }
  mr = blockReduceMax(mr);
  mc = blockReduceMax(mc);
  if (threadIdx.x==0) scale[0] = 1.f/(mr*mc);
}

__global__ void zinit(const float* __restrict__ a2, const float* __restrict__ scale, float* __restrict__ z){
  size_t idx = (size_t)blockIdx.x*blockDim.x + threadIdx.x;   // 16*256*256
  if (idx >= (size_t)16*65536) return;
  int j = idx & 255, i = (idx>>8) & 255, bh = (int)(idx>>16);
  z[idx] = a2[((size_t)bh<<16) + j*256 + i] * scale[0];
}

__global__ void conv_combine(const float* __restrict__ attn1, const __bf16* __restrict__ v,
                             const float* __restrict__ rw, __bf16* __restrict__ xo)
{
  size_t idx = (size_t)blockIdx.x*blockDim.x + threadIdx.x;   // 2*8*16640*64
  if (idx >= (size_t)17039360) return;
  int d = idx & 63;
  size_t r = idx >> 6;
  int t = (int)(r % 16640);
  size_t bh = r / 16640;
  int h = (int)(bh & 7), b = (int)(bh >> 3);
  float acc = attn1[idx];
  for (int j=0;j<33;j++){
    int tt = t + j - 16;
    if (tt >= 0 && tt < 16640)
      acc += rw[h*33+j] * (float)v[(bh*16640 + tt)*64 + d];
  }
  xo[((size_t)(b*16640 + t))*512 + h*64 + d] = (__bf16)acc;
}

__global__ void ppeg_kernel(const float* __restrict__ hin, float* __restrict__ hout,
    const float* __restrict__ w7, const float* __restrict__ b7,
    const float* __restrict__ w5, const float* __restrict__ b5,
    const float* __restrict__ w3, const float* __restrict__ b3)
{
  size_t idx = (size_t)blockIdx.x*blockDim.x + threadIdx.x;   // 2*16385*512
  if (idx >= (size_t)16778240) return;
  int c = idx % 512;
  size_t r = idx / 512;
  int t = (int)(r % 16385), b = (int)(r / 16385);
  if (t == 0){ hout[idx] = hin[idx]; return; }
  int y = (t-1)>>7, x = (t-1)&127;
  const float* f = hin + ((size_t)b*16385 + 1)*512 + c;
  float acc = f[((size_t)(y*128+x))*512] + b7[c] + b5[c] + b3[c];
  for (int dy=-3; dy<=3; dy++){
    int yy = y + dy; if (yy < 0 || yy >= 128) continue;
    for (int dx=-3; dx<=3; dx++){
      int xx = x + dx; if (xx < 0 || xx >= 128) continue;
      float pix = f[((size_t)(yy*128+xx))*512];
      acc += w7[c*49 + (dy+3)*7 + (dx+3)] * pix;
      if (dy>=-2 && dy<=2 && dx>=-2 && dx<=2) acc += w5[c*25 + (dy+2)*5 + (dx+2)] * pix;
      if (dy>=-1 && dy<=1 && dx>=-1 && dx<=1) acc += w3[c*9  + (dy+1)*3 + (dx+1)] * pix;
    }
  }
  hout[idx] = acc;
}

__global__ __launch_bounds__(256) void final_head(const float* __restrict__ h,
    const float* __restrict__ g, const float* __restrict__ beta,
    const float* __restrict__ cw, const float* __restrict__ cb, float* __restrict__ out)
{
  int b = blockIdx.x;
  const float* src = h + (size_t)b*16385*512;
  float s = 0.f;
  for (int c=threadIdx.x; c<512; c+=256) s += src[c];
  float mu = blockReduceSum(s) * (1.f/512.f);
  float v = 0.f;
  for (int c=threadIdx.x; c<512; c+=256){ float d = src[c]-mu; v += d*d; }
  float var = blockReduceSum(v) * (1.f/512.f);
  float rstd = rsqrtf(var + 1e-5f);
  float p0 = 0.f, p1 = 0.f;
  for (int c=threadIdx.x; c<512; c+=256){
    float nx = (src[c]-mu)*rstd*g[c] + beta[c];
    p0 += nx*cw[c*2]; p1 += nx*cw[c*2+1];
  }
  p0 = blockReduceSum(p0);
  p1 = blockReduceSum(p1);
  if (threadIdx.x==0){ out[b*2] = p0 + cb[0]; out[b*2+1] = p1 + cb[1]; }
}

// =============================== host side =================================
struct WS {
  float *hbuf, *hbuf2, *qlf, *klf, *a2, *z, *xz, *t3, *t4, *av, *attn1;
  float *accp, *mp, *lp, *rs, *cs, *scale;
  __bf16 *xpad, *q, *k, *v, *qlb, *klb, *wbuf, *xo;
};

static inline void* wsAlloc(char*& p, size_t bytes){
  void* r = p; p += (bytes + 255) & ~(size_t)255; return r;
}

static void run_attention(hipStream_t s, float* hbase,
    const float* g, const float* beta, const float* qkvW,
    const float* outW, const float* outB, const float* resW, const WS& w)
{
  ln_to_xpad<<<33280,256,0,s>>>(hbase, g, beta, w.xpad);
  gemm_big<1><<<dim3(260,12),256,0,s>>>(nullptr, w.xpad, qkvW, nullptr,
                                        33280, 512, 1536, nullptr, w.q, w.k, w.v);
  landmarks<<<1024,256,0,s>>>(w.q, w.k, w.qlf, w.klf, w.qlb, w.klb);
  // sim2 = ql @ kl^T  (NT), then softmax -> a2
  gemm_small_x3<true><<<dim3(4,4,16),256,0,s>>>(w.qlf, w.klf, w.a2, nullptr, nullptr,
      256,256,64, 16384,16384,65536, 1.f,0.f,0.f);
  softmax256<<<4096,256,0,s>>>(w.a2);
  colrow_sums<<<16,256,0,s>>>(w.a2, w.rs, w.cs);
  max_scale<<<1,256,0,s>>>(w.rs, w.cs, w.scale);
  zinit<<<4096,256,0,s>>>(w.a2, w.scale, w.z);
  // Newton-Schulz pinv, 6 iterations (bf16x3 compensated GEMMs)
  float* z  = w.z;
  float* t3 = w.t3;
  for (int it=0; it<6; ++it){
    gemm_small_x3<false><<<dim3(4,4,16),256,0,s>>>(w.a2, z, w.xz, nullptr, nullptr,
        256,256,256, 65536,65536,65536, 1.f,0.f,0.f);                 // xz = a2@z
    gemm_small_x3<false><<<dim3(4,4,16),256,0,s>>>(w.xz, w.xz, t3, nullptr, w.xz,
        256,256,256, 65536,65536,65536, 1.f,15.f,-7.f);               // t3 = xz@xz +15I -7xz
    gemm_small_x3<false><<<dim3(4,4,16),256,0,s>>>(w.xz, t3, w.t4, nullptr, nullptr,
        256,256,256, 65536,65536,65536, 1.f,0.f,0.f);                 // t4 = xz@t3
    gemm_small_x3<false><<<dim3(4,4,16),256,0,s>>>(z, w.t4, t3, nullptr, z,
        256,256,256, 65536,65536,65536, -0.25f,0.f,3.25f);            // z' = -0.25 z@t4 + 3.25 z
    float* tmp = z; z = t3; t3 = tmp;
  }
  // av = softmax(ql @ k^T) @ v, chunked flash + merge
  flash_kernel<<<dim3(1,26,16),256,0,s>>>(w.qlb, w.k, w.v, 256, 16640, 26,
                                          w.accp, w.mp, w.lp);
  flash_merge<<<1024,256,0,s>>>(w.accp, w.mp, w.lp, w.av, 256, 26);
  // wbuf = a2inv @ av  (bf16 out)
  gemm_small_x3<false><<<dim3(4,1,16),256,0,s>>>(z, w.av, w.xz, w.wbuf, nullptr,
      256,64,256, 65536,16384,16384, 1.f,0.f,0.f);
  // attn1 = softmax(q @ kl^T) @ wbuf   (single chunk flash)
  flash_kernel<<<dim3(65,1,16),256,0,s>>>(w.q, w.klb, w.wbuf, 16640, 256, 1,
                                          w.attn1, w.mp, w.lp);
  conv_combine<<<66560,256,0,s>>>(w.attn1, w.v, resW, w.xo);
  // hbase[:, -n:] += xo[255:] @ outW + outB
  gemm_big<2><<<dim3(257,4),256,0,s>>>(nullptr, w.xo, outW, outB,
                                       32770, 512, 512, hbase, nullptr, nullptr, nullptr);
}

extern "C" void kernel_launch(void* const* d_in, const int* in_sizes, int n_in,
                              void* d_out, int out_size, void* d_ws, size_t ws_size,
                              hipStream_t stream)
{
  (void)in_sizes; (void)n_in; (void)out_size; (void)ws_size;
  const float* x        = (const float*)d_in[0];
  const float* w_feat   = (const float*)d_in[1];
  const float* b_feat   = (const float*)d_in[2];
  const float* cls_tok  = (const float*)d_in[3];
  const float* ln1_g    = (const float*)d_in[4];
  const float* ln1_b    = (const float*)d_in[5];
  const float* qkv1     = (const float*)d_in[6];
  const float* out1_w   = (const float*)d_in[7];
  const float* out1_b   = (const float*)d_in[8];
  const float* res1_w   = (const float*)d_in[9];
  const float* ppeg_w7  = (const float*)d_in[10];
  const float* ppeg_b7  = (const float*)d_in[11];
  const float* ppeg_w5  = (const float*)d_in[12];
  const float* ppeg_b5  = (const float*)d_in[13];
  const float* ppeg_w3  = (const float*)d_in[14];
  const float* ppeg_b3  = (const float*)d_in[15];
  const float* ln2_g    = (const float*)d_in[16];
  const float* ln2_b    = (const float*)d_in[17];
  const float* qkv2     = (const float*)d_in[18];
  const float* out2_w   = (const float*)d_in[19];
  const float* out2_b   = (const float*)d_in[20];
  const float* res2_w   = (const float*)d_in[21];
  const float* lnf_g    = (const float*)d_in[22];
  const float* lnf_b    = (const float*)d_in[23];
  const float* cls_w    = (const float*)d_in[24];
  const float* cls_b    = (const float*)d_in[25];

  char* p = (char*)d_ws;
  WS w;
  w.hbuf  = (float*)wsAlloc(p, (size_t)2*16385*512*4);
  w.hbuf2 = (float*)wsAlloc(p, (size_t)2*16385*512*4);
  w.attn1 = (float*)wsAlloc(p, (size_t)2*8*16640*64*4);
  w.accp  = (float*)wsAlloc(p, (size_t)16*26*256*64*4);
  w.qlf   = (float*)wsAlloc(p, (size_t)16*256*64*4);
  w.klf   = (float*)wsAlloc(p, (size_t)16*256*64*4);
  w.a2    = (float*)wsAlloc(p, (size_t)16*256*256*4);
  w.z     = (float*)wsAlloc(p, (size_t)16*256*256*4);
  w.xz    = (float*)wsAlloc(p, (size_t)16*256*256*4);
  w.t3    = (float*)wsAlloc(p, (size_t)16*256*256*4);
  w.t4    = (float*)wsAlloc(p, (size_t)16*256*256*4);
  w.av    = (float*)wsAlloc(p, (size_t)16*256*64*4);
  w.mp    = (float*)wsAlloc(p, (size_t)16*26*256*4);
  w.lp    = (float*)wsAlloc(p, (size_t)16*26*256*4);
  w.rs    = (float*)wsAlloc(p, (size_t)16*256*4);
  w.cs    = (float*)wsAlloc(p, (size_t)16*256*4);
  w.scale = (float*)wsAlloc(p, 256);
  w.xpad  = (__bf16*)wsAlloc(p, (size_t)2*16640*512*2);
  w.q     = (__bf16*)wsAlloc(p, (size_t)2*8*16640*64*2);
  w.k     = (__bf16*)wsAlloc(p, (size_t)2*8*16640*64*2);
  w.v     = (__bf16*)wsAlloc(p, (size_t)2*8*16640*64*2);
  w.qlb   = (__bf16*)wsAlloc(p, (size_t)16*256*64*2);
  w.klb   = (__bf16*)wsAlloc(p, (size_t)16*256*64*2);
  w.wbuf  = (__bf16*)wsAlloc(p, (size_t)16*256*64*2);
  w.xo    = (__bf16*)wsAlloc(p, (size_t)2*16640*512*2);

  // h = [cls, gelu(x @ w_feat + b_feat)]
  cls_init<<<4,256,0,stream>>>(cls_tok, w.hbuf);
  gemm_big<0><<<dim3(256,4),256,0,stream>>>(x, nullptr, w_feat, b_feat,
                                            32768, 1024, 512, w.hbuf,
                                            nullptr, nullptr, nullptr);
  // h = h + attn1(LN1(h))
  run_attention(stream, w.hbuf, ln1_g, ln1_b, qkv1, out1_w, out1_b, res1_w, w);
  // h = PPEG(h)
  ppeg_kernel<<<65540,256,0,stream>>>(w.hbuf, w.hbuf2,
      ppeg_w7, ppeg_b7, ppeg_w5, ppeg_b5, ppeg_w3, ppeg_b3);
  // h = h + attn2(LN2(h))
  run_attention(stream, w.hbuf2, ln2_g, ln2_b, qkv2, out2_w, out2_b, res2_w, w);
  // logits = LNf(h)[:,0] @ cls_w + cls_b
  final_head<<<2,256,0,stream>>>(w.hbuf2, lnf_g, lnf_b, cls_w, cls_b, (float*)d_out);
}
